// PointNet_2267742732835
// MI455X (gfx1250) — compile-verified, matrix operand order checked by reading, not run on hardware
//
#include <hip/hip_runtime.h>
#include <hip/hip_bf16.h>

typedef __attribute__((ext_vector_type(16))) _Float16 v16h;
typedef __attribute__((ext_vector_type(8)))  _Float16 v8h;
typedef __attribute__((ext_vector_type(8)))  float    v8f;

#define DEVINL __device__ __forceinline__

// ---------------- problem dims ----------------
static constexpr int Mb   = 64;     // bboxes
static constexpr int Np   = 8192;   // points per bbox
static constexpr int INF  = 6;      // input features
static constexpr int HID  = 256;
static constexpr int SOUT = 128;
static constexpr int CDIM = INF + HID + SOUT;  // 390
static constexpr int TILE = 128;    // points per workgroup
static constexpr int KP1  = 32;     // layer-1 K padded 6 -> 32

// ---------------- workspace layout (bytes) ----------------
static constexpr size_t OFF_EW1T = 0;                          // [256][32]  f16 (W1^T, K padded)
static constexpr size_t OFF_EW2T = OFF_EW1T + 256*32*2;        // [256][256] f16
static constexpr size_t OFF_EW3T = OFF_EW2T + 256*256*2;       // [256][256] f16
static constexpr size_t OFF_SW1T = OFF_EW3T + 256*256*2;       // [128][256] f16
static constexpr size_t OFF_SW2T = OFF_SW1T + 128*256*2;       // [128][128] f16
static constexpr size_t OFF_SW3T = OFF_SW2T + 128*128*2;       // [128][128] f16
static constexpr size_t OFF_POOL = OFF_SW3T + 128*128*2;       // [64][256]  f32 (bits, max>=0)
static constexpr size_t OFF_SEG  = OFF_POOL + 64*256*4;        // [64][128]  f32
static constexpr size_t OFF_MINK = OFF_SEG  + 64*128*4;        // [64] u32 keys
static constexpr size_t OFF_MAXK = OFF_MINK + 64*4;            // [64] u32 keys
static constexpr size_t OFF_RAW  = OFF_MAXK + 64*4;            // [64][8192] f32 raw scores

// LDS halves: actA | actB | actC | wlds   (embed/classify kernels)
static constexpr int L_ACTA = 0;
static constexpr int L_ACTB = L_ACTA + TILE * KP1;       //  4096
static constexpr int L_ACTC = L_ACTB + TILE * HID;       // 36864
static constexpr int L_WLDS = L_ACTC + TILE * HID;       // 69632
static constexpr size_t SMEM_EMBED = (size_t)(L_WLDS + HID * HID) * 2;      // 270336 B

// LDS halves: a0 | a1 | a2 | wlds   (segment kernel)
static constexpr int S_A0   = 0;
static constexpr int S_A1   = S_A0 + 64 * HID;
static constexpr int S_A2   = S_A1 + 64 * SOUT;
static constexpr int S_WLDS = S_A2 + 64 * SOUT;
static constexpr size_t SMEM_SEG = (size_t)(S_WLDS + SOUT * HID) * 2;       // 131072 B

// order-preserving float <-> uint key (for global min/max atomics)
DEVINL unsigned fkey(float f) {
    unsigned u = __float_as_uint(f);
    return (u & 0x80000000u) ? ~u : (u | 0x80000000u);
}
DEVINL float funkey(unsigned k) {
    return __uint_as_float((k & 0x80000000u) ? (k & 0x7FFFFFFFu) : ~k);
}

// ---- XOR bank swizzle on 16-byte (8-half) chunks of a row-major [R][K] f16
// array in LDS (kills the 16-way bank conflict of 512/256-byte row strides).
// Returns offset in halves; each 8-half chunk stays contiguous.
template <int K>
DEVINL int swz(int r, int k) {
    constexpr int CPR = K / 8;                      // 16B chunks per row (pow2)
    constexpr int CM  = (CPR < 32 ? CPR : 32) - 1;  // swizzle mask
    return r * K + ((((k >> 3) ^ (r & CM)) & (CPR - 1)) << 3) + (k & 7);
}

// ---- WMMA fragment load from swizzled LDS tile: 16x32 f16 at (row0, kt*32).
// Lane l<16 holds row l, K = 0..7 / 16..23; lane l>=16 holds row l-16,
// K = 8..15 / 24..31 (CDNA5 16-bit A layout; the B slot takes the same
// pattern applied to B^T, i.e. an identical row-major 16x32 tile).
template <int K>
DEVINL v16h load_frag16(const _Float16* base, int row0, int kt) {
    const int lane = threadIdx.x & 31;
    const int r  = row0 + (lane & 15);
    const int k0 = kt * 32 + ((lane >> 4) & 1) * 8;
    const _Float16* p  = base + swz<K>(r, k0);
    const _Float16* p2 = base + swz<K>(r, k0 + 16);
    v16h f{};
#pragma unroll
    for (int i = 0; i < 8; ++i) { f[i] = p[i]; f[i + 8] = p2[i]; }
    return f;
}

// ---- cooperative copy of a [rows][K] f16 weight matrix global -> LDS with
// swizzle, 16B granules (global_load_b128 + conflict-free ds_store_b128)
template <int K>
DEVINL void stage_w(const _Float16* __restrict__ g, _Float16* lds, int rows) {
    constexpr int CPR = K / 8;
    constexpr int CM  = (CPR < 32 ? CPR : 32) - 1;
    const uint4* src = (const uint4*)g;
    uint4*       dst = (uint4*)lds;
    for (int i = threadIdx.x; i < rows * CPR; i += 256)
        dst[(i / CPR) * CPR + ((i % CPR) ^ ((i / CPR) & CM))] = src[i];
}

// One Linear(K->NOUT)+bias+ReLU layer, f16 in / f16 out, per-wave 16-point strip.
// Flipped orientation: D^T = Wt x H^T  (A slot = weight frag streamed from LDS,
// B slot = activation frag hoisted in VGPRs). Each lane's 8 accumulator
// elements are then 8 *consecutive output columns* of one point -> a single
// aligned ds_store_b128 per 16x16 tile. Weight frags are 2-deep software
// pipelined so every LDS load is covered by one WMMA.
template <int K, int NOUT>
DEVINL void mlp_layer_f16(const _Float16* actIn, const _Float16* Wlds,
                          const float* __restrict__ bias, _Float16* actOut,
                          int row0, int ct0, int nct) {
    constexpr int KT = K / 32;
    const int lane = threadIdx.x & 31;
    v16h bfrag[KT];                       // this wave's activation strip (invariant)
#pragma unroll
    for (int kt = 0; kt < KT; ++kt)
        bfrag[kt] = load_frag16<K>(actIn, row0, kt);
    const int hi = (lane >> 4) & 1;
    const int pt = row0 + (lane & 15);
    for (int ct = ct0; ct < ct0 + nct; ++ct) {
        v8f acc = {};
        v16h wcur = load_frag16<K>(Wlds, ct * 16, 0);
#pragma unroll
        for (int kt = 0; kt < KT; ++kt) {
            v16h wnext = wcur;
            if (kt + 1 < KT) wnext = load_frag16<K>(Wlds, ct * 16, kt + 1);
            acc = __builtin_amdgcn_wmma_f32_16x16x32_f16(false, wcur, false, bfrag[kt],
                                                         (short)0, acc, false, false);
            wcur = wnext;
        }
        // D^T layout: acc[i] -> outcol = ct*16 + hi*8 + i, point = pt
        const int col0 = ct * 16 + hi * 8;
        const float4* bp = (const float4*)(bias + col0);
        float4 b0 = bp[0], b1 = bp[1];
        float bb[8] = {b0.x, b0.y, b0.z, b0.w, b1.x, b1.y, b1.z, b1.w};
        union { v8h h; uint4 u; } o;
#pragma unroll
        for (int i = 0; i < 8; ++i) {
            float v = acc[i] + bb[i];
            o.h[i] = (_Float16)(v > 0.f ? v : 0.f);
        }
        *(uint4*)(actOut + swz<NOUT>(pt, col0)) = o.u;   // one chunk, 16B aligned
    }
}

// Full 3-layer embed MLP for a 128-point tile; result (swizzled 128x256 f16)
// left in actB. Weights staged layer-by-layer into wlds (max 128 KB).
DEVINL void embed_tile(const float* __restrict__ feat, int m, int pt0,
                       const _Float16* __restrict__ eW1t, const float* __restrict__ eb1,
                       const _Float16* __restrict__ eW2t, const float* __restrict__ eb2,
                       const _Float16* __restrict__ eW3t, const float* __restrict__ eb3,
                       _Float16* actA, _Float16* actB, _Float16* actC, _Float16* wlds) {
    const int tid = threadIdx.x;
    __builtin_prefetch(eW2t + tid * 64, 0, 1);   // global_prefetch_b8: warm weight stream
    stage_w<KP1>(eW1t, wlds, HID);               // 16 KB
    for (int i = tid; i < TILE * KP1; i += 256) {
        int r = i >> 5, k = i & 31;
        float v = (k < INF) ? feat[((size_t)m * Np + pt0 + r) * INF + k] : 0.f;
        actA[swz<KP1>(r, k)] = (_Float16)v;
    }
    __syncthreads();
    const int wave = tid >> 5;
    const int row0 = wave * 16;                  // 8 waves x 16 points = 128
    mlp_layer_f16<KP1, HID>(actA, wlds, eb1, actB, row0, 0, HID / 16);
    __syncthreads();
    stage_w<HID>(eW2t, wlds, HID);               // 128 KB
    __syncthreads();
    mlp_layer_f16<HID, HID>(actB, wlds, eb2, actC, row0, 0, HID / 16);
    __syncthreads();
    stage_w<HID>(eW3t, wlds, HID);
    __syncthreads();
    mlp_layer_f16<HID, HID>(actC, wlds, eb3, actB, row0, 0, HID / 16);
    __syncthreads();
}

// ---------------- kernels ----------------

// transpose + f16 convert: W[K][N] f32 -> Wt[N][Kpad] f16 (zero pad K)
__global__ void wconv_kernel(const float* __restrict__ W, _Float16* __restrict__ Wt,
                             int K, int Kpad, int Nn) {
    int i = blockIdx.x * blockDim.x + threadIdx.x;
    if (i >= Nn * Kpad) return;
    int n = i / Kpad, k = i % Kpad;
    Wt[i] = (k < K) ? (_Float16)W[(size_t)k * Nn + n] : (_Float16)0.0f;
}

__global__ void init_kernel(float* pooled, unsigned* minkey, unsigned* maxkey) {
    int i = blockIdx.x * blockDim.x + threadIdx.x;
    if (i < Mb * HID) pooled[i] = 0.f;
    if (i < Mb) { minkey[i] = 0xFFFFFFFFu; maxkey[i] = 0u; }
}

__global__ __launch_bounds__(256) void embed_pool_kernel(
        const float* __restrict__ feat,
        const _Float16* __restrict__ eW1t, const float* __restrict__ eb1,
        const _Float16* __restrict__ eW2t, const float* __restrict__ eb2,
        const _Float16* __restrict__ eW3t, const float* __restrict__ eb3,
        float* __restrict__ pooled) {
    extern __shared__ _Float16 smem[];
    const int m   = blockIdx.x >> 6;          // 64 tiles per bbox
    const int pt0 = (blockIdx.x & 63) * TILE;
    embed_tile(feat, m, pt0, eW1t, eb1, eW2t, eb2, eW3t, eb3,
               smem + L_ACTA, smem + L_ACTB, smem + L_ACTC, smem + L_WLDS);
    // per-column max over 128 rows, then global atomic-max (values >= 0 post-ReLU,
    // so raw-bit unsigned max is order-preserving and exact)
    const int tid = threadIdx.x;              // col 0..255
    const _Float16* actB = smem + L_ACTB;
    float vmax = 0.f;
    for (int r = 0; r < TILE; ++r)
        vmax = fmaxf(vmax, (float)actB[swz<HID>(r, tid)]);
    atomicMax((unsigned*)pooled + (size_t)m * HID + tid, __float_as_uint(vmax));
}

__global__ __launch_bounds__(256) void segment_kernel(
        const float* __restrict__ pooled,
        const _Float16* __restrict__ sW1t, const float* __restrict__ sb1,
        const _Float16* __restrict__ sW2t, const float* __restrict__ sb2,
        const _Float16* __restrict__ sW3t, const float* __restrict__ sb3,
        float* __restrict__ seg) {
    extern __shared__ _Float16 smem[];
    _Float16* a0   = smem + S_A0;
    _Float16* a1   = smem + S_A1;
    _Float16* a2   = smem + S_A2;
    _Float16* wlds = smem + S_WLDS;
    const int tid = threadIdx.x;
    for (int i = tid; i < Mb * HID; i += 256)
        a0[swz<HID>(i >> 8, i & 255)] = (_Float16)pooled[i];
    stage_w<HID>(sW1t, wlds, SOUT);           // 64 KB
    __syncthreads();
    const int wave = tid >> 5;
    const int row0 = (wave & 3) * 16;           // 4 row tiles of 64 rows
    const int ct0  = (wave >> 2) * (SOUT / 32); // split 8 col tiles over 2 wave groups
    mlp_layer_f16<HID, SOUT>(a0, wlds, sb1, a1, row0, ct0, SOUT / 32);
    __syncthreads();
    stage_w<SOUT>(sW2t, wlds, SOUT);
    __syncthreads();
    mlp_layer_f16<SOUT, SOUT>(a1, wlds, sb2, a2, row0, ct0, SOUT / 32);
    __syncthreads();
    stage_w<SOUT>(sW3t, wlds, SOUT);
    __syncthreads();
    mlp_layer_f16<SOUT, SOUT>(a2, wlds, sb3, a1, row0, ct0, SOUT / 32);
    __syncthreads();
    for (int i = tid; i < Mb * SOUT; i += 256)
        seg[i] = (float)a1[swz<SOUT>(i >> 7, i & 127)];
}

__global__ __launch_bounds__(256) void classify_kernel(
        const float* __restrict__ feat,
        const _Float16* __restrict__ eW1t, const float* __restrict__ eb1,
        const _Float16* __restrict__ eW2t, const float* __restrict__ eb2,
        const _Float16* __restrict__ eW3t, const float* __restrict__ eb3,
        const float* __restrict__ seg,
        const float* __restrict__ cW, const float* __restrict__ cb,
        float* __restrict__ raw, unsigned* __restrict__ minkey, unsigned* __restrict__ maxkey) {
    extern __shared__ _Float16 smem[];
    __shared__ float s_cw[CDIM + 2];
    __shared__ float s_seg[SOUT];
    __shared__ float s_sc[TILE];
    const int m   = blockIdx.x >> 6;
    const int pt0 = (blockIdx.x & 63) * TILE;
    const int tid = threadIdx.x;
    // recompute hidden for this tile (cheaper than round-tripping 256 MB to HBM)
    embed_tile(feat, m, pt0, eW1t, eb1, eW2t, eb2, eW3t, eb3,
               smem + L_ACTA, smem + L_ACTB, smem + L_ACTC, smem + L_WLDS);
    for (int i = tid; i < CDIM; i += 256) s_cw[i] = cW[i];
    if (tid == 0) s_cw[CDIM] = cb[0];
    for (int i = tid; i < SOUT; i += 256) s_seg[i] = seg[(size_t)m * SOUT + i];
    __syncthreads();
    const _Float16* actB = smem + L_ACTB;
    if (tid < TILE) {
        const int p = pt0 + tid;
        const float* f = &feat[((size_t)m * Np + p) * INF];
        float s = s_cw[CDIM];
#pragma unroll
        for (int j = 0; j < INF; ++j) s += f[j] * s_cw[j];
        // chunk-wise swizzled read: one address calc + ds_load_b128 per 8 halves
        for (int c = 0; c < HID / 8; ++c) {
            const v8h hv = *(const v8h*)(actB + (size_t)tid * HID + ((c ^ (tid & 31)) << 3));
#pragma unroll
            for (int e = 0; e < 8; ++e)
                s += (float)hv[e] * s_cw[INF + c * 8 + e];
        }
        for (int j = 0; j < SOUT; ++j) s += s_seg[j] * s_cw[INF + HID + j];
        raw[(size_t)m * Np + p] = s;
        s_sc[tid] = s;
    }
    __syncthreads();
    if (tid == 0) {
        float mn = s_sc[0], mx = s_sc[0];
        for (int i = 1; i < TILE; ++i) { mn = fminf(mn, s_sc[i]); mx = fmaxf(mx, s_sc[i]); }
        atomicMin(&minkey[m], fkey(mn));
        atomicMax(&maxkey[m], fkey(mx));
    }
}

__global__ __launch_bounds__(256) void rescale_kernel(
        const float* __restrict__ raw, const unsigned* __restrict__ minkey,
        const unsigned* __restrict__ maxkey, float* __restrict__ out) {
    int i = blockIdx.x * blockDim.x + threadIdx.x;
    if (i >= Mb * Np) return;
    int m = i >> 13;  // /8192
    float mn = funkey(minkey[m]), mx = funkey(maxkey[m]);
    out[i] = (raw[i] - mn) / (mx - mn) * 2.f - 1.f;
}

// ---------------- host launcher ----------------
extern "C" void kernel_launch(void* const* d_in, const int* in_sizes, int n_in,
                              void* d_out, int out_size, void* d_ws, size_t ws_size,
                              hipStream_t stream) {
    const float* feat = (const float*)d_in[0];
    const float* eW1 = (const float*)d_in[1];  const float* eb1 = (const float*)d_in[2];
    const float* eW2 = (const float*)d_in[3];  const float* eb2 = (const float*)d_in[4];
    const float* eW3 = (const float*)d_in[5];  const float* eb3 = (const float*)d_in[6];
    const float* sW1 = (const float*)d_in[7];  const float* sb1 = (const float*)d_in[8];
    const float* sW2 = (const float*)d_in[9];  const float* sb2 = (const float*)d_in[10];
    const float* sW3 = (const float*)d_in[11]; const float* sb3 = (const float*)d_in[12];
    const float* cW  = (const float*)d_in[13]; const float* cb  = (const float*)d_in[14];

    char* ws = (char*)d_ws;
    _Float16* eW1t = (_Float16*)(ws + OFF_EW1T);
    _Float16* eW2t = (_Float16*)(ws + OFF_EW2T);
    _Float16* eW3t = (_Float16*)(ws + OFF_EW3T);
    _Float16* sW1t = (_Float16*)(ws + OFF_SW1T);
    _Float16* sW2t = (_Float16*)(ws + OFF_SW2T);
    _Float16* sW3t = (_Float16*)(ws + OFF_SW3T);
    float*    pooled = (float*)(ws + OFF_POOL);
    float*    seg    = (float*)(ws + OFF_SEG);
    unsigned* minkey = (unsigned*)(ws + OFF_MINK);
    unsigned* maxkey = (unsigned*)(ws + OFF_MAXK);
    float*    raw    = (float*)(ws + OFF_RAW);

    auto cdiv = [](int a, int b) { return (a + b - 1) / b; };

    // prep: weight transpose/convert + init accumulators
    wconv_kernel<<<cdiv(256 * 32, 256),  256, 0, stream>>>(eW1, eW1t, INF, KP1, HID);
    wconv_kernel<<<cdiv(256 * 256, 256), 256, 0, stream>>>(eW2, eW2t, HID, HID, HID);
    wconv_kernel<<<cdiv(256 * 256, 256), 256, 0, stream>>>(eW3, eW3t, HID, HID, HID);
    wconv_kernel<<<cdiv(128 * 256, 256), 256, 0, stream>>>(sW1, sW1t, HID, HID, SOUT);
    wconv_kernel<<<cdiv(128 * 128, 256), 256, 0, stream>>>(sW2, sW2t, SOUT, SOUT, SOUT);
    wconv_kernel<<<cdiv(128 * 128, 256), 256, 0, stream>>>(sW3, sW3t, SOUT, SOUT, SOUT);
    init_kernel<<<cdiv(Mb * HID, 256), 256, 0, stream>>>(pooled, minkey, maxkey);

    const int nTiles = Mb * (Np / TILE);  // 4096 blocks
    embed_pool_kernel<<<nTiles, 256, SMEM_EMBED, stream>>>(
        feat, eW1t, eb1, eW2t, eb2, eW3t, eb3, pooled);
    segment_kernel<<<1, 256, SMEM_SEG, stream>>>(
        pooled, sW1t, sb1, sW2t, sb2, sW3t, sb3, seg);
    classify_kernel<<<nTiles, 256, SMEM_EMBED, stream>>>(
        feat, eW1t, eb1, eW2t, eb2, eW3t, eb3, seg, cW, cb, raw, minkey, maxkey);
    rescale_kernel<<<cdiv(Mb * Np, 256), 256, 0, stream>>>(raw, minkey, maxkey, (float*)d_out);
}
